// MultiHeadAttention_44195213475961
// MI455X (gfx1250) — matrix operand order repaired and checked by reading, not compiled, not run on hardware
//
#include <hip/hip_runtime.h>

// ---------------------------------------------------------------------------
// MHA forward for B=2, S=2048, D=768, H=12, dk=64 on gfx1250 (MI455X).
// Pipeline:
//   1) qkv_proj_kernel : X @ W + b for Q,K,V (fp32 in -> f16 out), writing
//      Q,K as [B,H,S,dk] and V transposed as [B,H,dk,S] (WMMA-frag friendly).
//   2) attn_kernel     : flash-style causal attention, f16 WMMA, fp32 accum,
//      online softmax; K/V tiles double-buffered in LDS via
//      global_load_async_to_lds_b128 (ASYNCcnt pipeline); ctx f16 [B,S,D].
//   3) out_proj_kernel : ctx @ W_o + b_o -> fp32 output.
// All matmuls use v_wmma_f32_16x16x32_f16.
// ---------------------------------------------------------------------------

typedef __attribute__((ext_vector_type(16))) _Float16 v16h;
typedef __attribute__((ext_vector_type(8)))  _Float16 v8h;
typedef __attribute__((ext_vector_type(8)))  float    v8f;

#define NB 2
#define NS 2048
#define ND 768
#define NH 12
#define NDK 64
#define NM (NB * NS)          // 4096 rows in the projection GEMMs

__device__ __forceinline__ v8f wmma_f16(v16h a, v16h b, v8f c) {
  return __builtin_amdgcn_wmma_f32_16x16x32_f16(
      /*neg_a=*/false, a, /*neg_b=*/false, b,
      /*c_mod=*/(short)0, c, /*reuse_a=*/false, /*reuse_b=*/false);
}

__device__ __forceinline__ v16h make_v16(v8h lo, v8h hi) {
  v16h r;
#pragma unroll
  for (int i = 0; i < 8; ++i) { r[i] = lo[i]; r[i + 8] = hi[i]; }
  return r;
}

// A-matrix fragment (16x32 f16). ISA layout: lane<16 holds K={0..7,16..23},
// lane>=16 holds K={8..15,24..31}, row M = lane&15.
__device__ __forceinline__ v16h load_a_h(const _Float16* base, int row, int ld,
                                         int k0, int lane) {
  const _Float16* p = base + row * ld + k0 + ((lane & 16) ? 8 : 0);
  v8h lo = *(const v8h*)p;
  v8h hi = *(const v8h*)(p + 16);
  return make_v16(lo, hi);
}

// Same fragment built from fp32 source (converted to f16 on the fly).
__device__ __forceinline__ v16h load_a_f32(const float* base, int row, int ld,
                                           int k0, int lane) {
  const float* p = base + row * ld + k0 + ((lane & 16) ? 8 : 0);
  v16h r;
#pragma unroll
  for (int j = 0; j < 8; ++j) {
    r[j]     = (_Float16)p[j];
    r[j + 8] = (_Float16)p[16 + j];
  }
  return r;
}

// B-matrix fragment (32x16 f16). Column N = lane&15; lane<16 holds K={0..7,16..23},
// lane>=16 holds K={8..15,24..31}. Source stored [N][K] row-major with leading dim ld.
__device__ __forceinline__ v16h load_b_h(const _Float16* base, int n, int ld,
                                         int k0, int lane) {
  const _Float16* p = base + n * ld + k0 + ((lane & 16) ? 8 : 0);
  v8h lo = *(const v8h*)p;
  v8h hi = *(const v8h*)(p + 16);
  return make_v16(lo, hi);
}

// ---- gfx1250 async copy helpers (ASYNCcnt path) ---------------------------
__device__ __forceinline__ unsigned lds_off32(const void* p) {
  // Generic pointers to LDS carry the workgroup-relative offset in the low
  // 32 bits (aperture in the high bits) -- see CDNA5 ISA 10.2 aperture rules.
  return (unsigned)(unsigned long long)p;
}

__device__ __forceinline__ void async_ld_b128(unsigned ldsOff, const void* g) {
  asm volatile("global_load_async_to_lds_b128 %0, %1, off"
               :
               : "v"(ldsOff), "v"((unsigned long long)g)
               : "memory");
}

__device__ __forceinline__ void wait_async0() {
  asm volatile("s_wait_asynccnt 0x0" ::: "memory");
}

// ---------------------------------------------------------------------------
// Kernel 1: fused QKV projection.  grid = (M/64, N/64, 3), block = 128 (4 waves)
// ---------------------------------------------------------------------------
__global__ __launch_bounds__(128) void qkv_proj_kernel(
    const float* __restrict__ q_in, const float* __restrict__ k_in,
    const float* __restrict__ v_in,
    const float* __restrict__ Wq, const float* __restrict__ bq,
    const float* __restrict__ Wk, const float* __restrict__ bk,
    const float* __restrict__ Wv, const float* __restrict__ bv,
    _Float16* __restrict__ Qh, _Float16* __restrict__ Kh,
    _Float16* __restrict__ Vt) {
  const int proj = blockIdx.z;
  const float* X    = (proj == 0) ? q_in : (proj == 1) ? k_in : v_in;
  const float* W    = (proj == 0) ? Wq   : (proj == 1) ? Wk   : Wv;
  const float* bias = (proj == 0) ? bq   : (proj == 1) ? bk   : bv;

  const int m0 = blockIdx.x * 64;
  const int n0 = blockIdx.y * 64;
  const int tid = threadIdx.x;
  const int lane = tid & 31;
  const int wv = tid >> 5;
  const int col = lane & 15;
  const int rhalf = (lane & 16) ? 8 : 0;

  __shared__ _Float16 Bt[64 * 40];   // W tile, transposed: [n][k], padded ld=40

  v8f acc[4];
#pragma unroll
  for (int t = 0; t < 4; ++t) acc[t] = (v8f){};

  for (int k0 = 0; k0 < ND; k0 += 32) {
#pragma unroll
    for (int i = 0; i < 16; ++i) {
      int idx = i * 128 + tid;
      int kk = idx >> 6;
      int nn = idx & 63;
      Bt[nn * 40 + kk] = (_Float16)W[(k0 + kk) * ND + n0 + nn];
    }
    __syncthreads();

    v16h a = load_a_f32(X, m0 + wv * 16 + col, ND, k0, lane);
#pragma unroll
    for (int nt = 0; nt < 4; ++nt) {
      v16h b = load_b_h(Bt, nt * 16 + col, 40, 0, lane);
      acc[nt] = wmma_f16(a, b, acc[nt]);
    }
    __syncthreads();
  }

#pragma unroll
  for (int nt = 0; nt < 4; ++nt) {
#pragma unroll
    for (int r = 0; r < 8; ++r) {
      int m = m0 + wv * 16 + rhalf + r;
      int n = n0 + nt * 16 + col;
      float val = acc[nt][r] + bias[n];
      _Float16 hv = (_Float16)val;
      int b  = m >> 11;
      int s  = m & (NS - 1);
      int hh = n >> 6;
      int dk = n & (NDK - 1);
      if (proj == 2) {
        Vt[(((b * NH + hh) * NDK + dk) * NS) + s] = hv;      // [B,H,dk,S]
      } else {
        _Float16* dst = (proj == 0) ? Qh : Kh;
        dst[(((b * NH + hh) * NS + s) * NDK) + dk] = hv;     // [B,H,S,dk]
      }
    }
  }
}

// ---------------------------------------------------------------------------
// Kernel 2: flash-style causal attention with async-to-LDS K/V pipeline.
// grid = (S/64, H, B), block = 128 (4 waves). Wave w owns 16 query rows.
// ---------------------------------------------------------------------------
__global__ __launch_bounds__(128) void attn_kernel(
    const _Float16* __restrict__ Qh, const _Float16* __restrict__ Kh,
    const _Float16* __restrict__ Vt, _Float16* __restrict__ Ctx) {
  const int Q0 = blockIdx.x * 64;
  const int h  = blockIdx.y;
  const int bz = blockIdx.z;
  const int tid = threadIdx.x;
  const int lane = tid & 31;
  const int wv = tid >> 5;
  const int col = lane & 15;
  const int rhalf = (lane & 16) ? 8 : 0;
  const int bh = bz * NH + h;

  const _Float16* Qb = Qh + (size_t)bh * NS * NDK;
  const _Float16* Kb = Kh + (size_t)bh * NS * NDK;
  const _Float16* Vb = Vt + (size_t)bh * NDK * NS;

  __shared__ _Float16 Ktile[2][32 * 64];   // [key][dk]  (4 KB per buffer)
  __shared__ _Float16 Vtile[2][64 * 32];   // [dk][key]  (4 KB per buffer)
  __shared__ _Float16 Pt[4][16 * 40];      // per-wave P tile, padded ld=40
  _Float16* P = &Pt[wv][0];

  const int qb = Q0 + wv * 16;

  // Q fragments (dk halves), loaded once from global.
  v16h qa0 = load_a_h(Qb, qb + col, NDK, 0, lane);
  v16h qa1 = load_a_h(Qb, qb + col, NDK, 32, lane);

  float mi[8], li[8];
#pragma unroll
  for (int r = 0; r < 8; ++r) { mi[r] = -1e30f; li[r] = 0.0f; }
  v8f acc[4];
#pragma unroll
  for (int t = 0; t < 4; ++t) acc[t] = (v8f){};

  // Issue async copy of K/V tile `kb` into buffer `buf`.
  // K tile: 4 KB contiguous in global ([S,dk] rows). V tile: 64 rows x 64 B.
  auto issue_tile = [&](int buf, int kb) {
#pragma unroll
    for (int i = 0; i < 2; ++i) {
      int idx = i * 128 + tid;                 // 256 chunks of 8 f16
      async_ld_b128(lds_off32(&Ktile[buf][idx * 8]),
                    Kb + (size_t)kb * NDK + idx * 8);
    }
#pragma unroll
    for (int i = 0; i < 2; ++i) {
      int idx = i * 128 + tid;
      int row = idx >> 2;                      // dk row 0..63
      int ch  = idx & 3;                       // 4 chunks of 8 keys
      async_ld_b128(lds_off32(&Vtile[buf][row * 32 + ch * 8]),
                    Vb + (size_t)row * NS + kb + ch * 8);
    }
  };

  const int nkt = (Q0 + 64) / 32;       // causal bound, uniform per block
  issue_tile(0, 0);
  wait_async0();
  __syncthreads();

  for (int kt = 0; kt < nkt; ++kt) {
    const int kb = kt * 32;
    const int buf = kt & 1;
    if (kt + 1 < nkt) issue_tile(buf ^ 1, kb + 32);

    const _Float16* Kl = &Ktile[buf][0];
    const _Float16* Vl = &Vtile[buf][0];

    // ---- scores: S = Q(16x64) x K^T(64x32), two 16-key subtiles ----
    v8f s0 = (v8f){};
    v8f s1 = (v8f){};
    {
      v16h b0a = load_b_h(Kl, col, NDK, 0, lane);
      v16h b0b = load_b_h(Kl, col, NDK, 32, lane);
      s0 = wmma_f16(qa0, b0a, s0);
      s0 = wmma_f16(qa1, b0b, s0);
      v16h b1a = load_b_h(Kl, 16 + col, NDK, 0, lane);
      v16h b1b = load_b_h(Kl, 16 + col, NDK, 32, lane);
      s1 = wmma_f16(qa0, b1a, s1);
      s1 = wmma_f16(qa1, b1b, s1);
    }

    // ---- causal mask + online softmax update ----
#pragma unroll
    for (int r = 0; r < 8; ++r) {
      int q = qb + rhalf + r;
      if (kb + col      > q) s0[r] = -1e30f;
      if (kb + 16 + col > q) s1[r] = -1e30f;
    }
#pragma unroll
    for (int r = 0; r < 8; ++r) {
      float mx = fmaxf(s0[r], s1[r]);
      mx = fmaxf(mx, __shfl_xor(mx, 1, 32));
      mx = fmaxf(mx, __shfl_xor(mx, 2, 32));
      mx = fmaxf(mx, __shfl_xor(mx, 4, 32));
      mx = fmaxf(mx, __shfl_xor(mx, 8, 32));
      float mnew  = fmaxf(mi[r], mx);
      float scale = __expf(mi[r] - mnew);
      float p0 = __expf(s0[r] - mnew);
      float p1 = __expf(s1[r] - mnew);
      s0[r] = p0;
      s1[r] = p1;
      float rs = p0 + p1;
      rs += __shfl_xor(rs, 1, 32);
      rs += __shfl_xor(rs, 2, 32);
      rs += __shfl_xor(rs, 4, 32);
      rs += __shfl_xor(rs, 8, 32);
      li[r] = li[r] * scale + rs;
      mi[r] = mnew;
#pragma unroll
      for (int t = 0; t < 4; ++t) acc[t][r] *= scale;
    }

    // ---- P (C-frag layout) -> LDS -> A-frag layout ----
#pragma unroll
    for (int r = 0; r < 8; ++r) {
      P[(rhalf + r) * 40 + col]      = (_Float16)s0[r];
      P[(rhalf + r) * 40 + 16 + col] = (_Float16)s1[r];
    }
    __syncthreads();
    v16h pa = load_a_h(P, col, 40, 0, lane);

    // ---- O += P(16x32) x V(32x64) from LDS V tile ----
#pragma unroll
    for (int t = 0; t < 4; ++t) {
      v16h bv = load_b_h(Vl, t * 16 + col, 32, 0, lane);
      acc[t] = wmma_f16(pa, bv, acc[t]);
    }

    // End of tile: next tile's async copies must have landed, and every wave
    // must be done with buffer `buf` before iteration kt+1 re-issues into it.
    wait_async0();
    __syncthreads();
  }

  // ---- epilogue: normalize, write ctx f16 as [B,S,D] ----
#pragma unroll
  for (int t = 0; t < 4; ++t) {
#pragma unroll
    for (int r = 0; r < 8; ++r) {
      int q = qb + rhalf + r;
      float o = acc[t][r] * (1.0f / li[r]);
      Ctx[(size_t)(bz * NS + q) * ND + h * NDK + t * 16 + col] = (_Float16)o;
    }
  }
}

// ---------------------------------------------------------------------------
// Kernel 3: output projection. grid = (M/64, N/64), block = 128 (4 waves).
// ---------------------------------------------------------------------------
__global__ __launch_bounds__(128) void out_proj_kernel(
    const _Float16* __restrict__ Ctx, const float* __restrict__ Wo,
    const float* __restrict__ bo, float* __restrict__ Out) {
  const int m0 = blockIdx.x * 64;
  const int n0 = blockIdx.y * 64;
  const int tid = threadIdx.x;
  const int lane = tid & 31;
  const int wv = tid >> 5;
  const int col = lane & 15;
  const int rhalf = (lane & 16) ? 8 : 0;

  __shared__ _Float16 Bt[64 * 40];

  v8f acc[4];
#pragma unroll
  for (int t = 0; t < 4; ++t) acc[t] = (v8f){};

  for (int k0 = 0; k0 < ND; k0 += 32) {
#pragma unroll
    for (int i = 0; i < 16; ++i) {
      int idx = i * 128 + tid;
      int kk = idx >> 6;
      int nn = idx & 63;
      Bt[nn * 40 + kk] = (_Float16)Wo[(k0 + kk) * ND + n0 + nn];
    }
    __syncthreads();

    v16h a = load_a_h(Ctx, m0 + wv * 16 + col, ND, k0, lane);
#pragma unroll
    for (int nt = 0; nt < 4; ++nt) {
      v16h b = load_b_h(Bt, nt * 16 + col, 40, 0, lane);
      acc[nt] = wmma_f16(a, b, acc[nt]);
    }
    __syncthreads();
  }

#pragma unroll
  for (int nt = 0; nt < 4; ++nt) {
#pragma unroll
    for (int r = 0; r < 8; ++r) {
      int m = m0 + wv * 16 + rhalf + r;
      int n = n0 + nt * 16 + col;
      Out[(size_t)m * ND + n] = acc[nt][r] + bo[n];
    }
  }
}

// ---------------------------------------------------------------------------
extern "C" void kernel_launch(void* const* d_in, const int* in_sizes, int n_in,
                              void* d_out, int out_size, void* d_ws,
                              size_t ws_size, hipStream_t stream) {
  (void)in_sizes; (void)n_in; (void)out_size; (void)ws_size;

  const float* query = (const float*)d_in[0];
  const float* key_  = (const float*)d_in[1];
  const float* value = (const float*)d_in[2];
  // d_in[3] is the causal mask; causality is applied analytically in-kernel.
  const float* W_q = (const float*)d_in[4];
  const float* b_q = (const float*)d_in[5];
  const float* W_k = (const float*)d_in[6];
  const float* b_k = (const float*)d_in[7];
  const float* W_v = (const float*)d_in[8];
  const float* b_v = (const float*)d_in[9];
  const float* W_o = (const float*)d_in[10];
  const float* b_o = (const float*)d_in[11];

  const size_t perTensor = (size_t)NB * NH * NS * NDK;   // 3,145,728 elems
  _Float16* Qh  = (_Float16*)d_ws;
  _Float16* Kh  = Qh + perTensor;
  _Float16* Vt  = Kh + perTensor;
  _Float16* Ctx = Vt + perTensor;

  qkv_proj_kernel<<<dim3(NM / 64, ND / 64, 3), 128, 0, stream>>>(
      query, key_, value, W_q, b_q, W_k, b_k, W_v, b_v, Qh, Kh, Vt);

  attn_kernel<<<dim3(NS / 64, NH, NB), 128, 0, stream>>>(Qh, Kh, Vt, Ctx);

  out_proj_kernel<<<dim3(NM / 64, ND / 64), 128, 0, stream>>>(
      Ctx, W_o, b_o, (float*)d_out);
}
